// CentroidDistance_5763846111486
// MI455X (gfx1250) — matrix operand order repaired and checked by reading, not compiled, not run on hardware
//
#include <hip/hip_runtime.h>
#include <hip/hip_bf16.h>

#define N_NODES      262144
#define NUM_CENTROID 256
#define EMBED_DIM    128
#define NUM_GRAPHS   128
#define NSEG         8          // LDS segment accumulator depth (block spans ~1-2 graphs)

typedef __attribute__((ext_vector_type(16))) __bf16 v16bf;
typedef __attribute__((ext_vector_type(8)))  __bf16 v8bf;
typedef __attribute__((ext_vector_type(8)))  float  v8f;

// ---- workspace layout (bytes) ----
// [0,      65536)  centroids bf16 (256x128)
// [65536,  66560)  c_sq   (256 f32)
// [66560, 197632)  sums   (128x256 f32)
// [197632,198144)  counts (128 f32)
#define WS_CSQ    65536
#define WS_SUMS   66560
#define WS_COUNTS 197632

// -------- zero sums + counts --------
__global__ __launch_bounds__(256) void cd_zero_kernel(float* __restrict__ sums,
                                                      float* __restrict__ counts) {
  int i = blockIdx.x * 256 + threadIdx.x;
  if (i < NUM_GRAPHS * NUM_CENTROID) sums[i] = 0.f;
  if (i < NUM_GRAPHS) counts[i] = 0.f;
}

// -------- centroid f32 -> bf16 + row norms --------
__global__ __launch_bounds__(128) void cd_cent_kernel(const float* __restrict__ w,
                                                      unsigned short* __restrict__ centBf,
                                                      float* __restrict__ cSq) {
  __shared__ float red[128];
  int c = blockIdx.x, d = threadIdx.x;
  float v = w[c * EMBED_DIM + d];
  __bf16 b = (__bf16)v;
  centBf[c * EMBED_DIM + d] = *(unsigned short*)&b;
  red[d] = v * v;
  __syncthreads();
  #pragma unroll
  for (int s = 64; s > 0; s >>= 1) {
    if (d < s) red[d] += red[d + s];
    __syncthreads();
  }
  if (d == 0) cSq[c] = red[0];
}

// -------- per-graph node counts (LDS histogram) --------
__global__ __launch_bounds__(256) void cd_hist_kernel(const int* __restrict__ batch,
                                                      float* __restrict__ counts) {
  __shared__ float h[NUM_GRAPHS];
  int tid = threadIdx.x;
  if (tid < NUM_GRAPHS) h[tid] = 0.f;
  __syncthreads();
  int base = blockIdx.x * 2048;  // 128 blocks x 2048 nodes
  #pragma unroll
  for (int i = 0; i < 8; ++i) {
    int g = batch[base + i * 256 + tid];
    atomicAdd(&h[g], 1.f);
  }
  __syncthreads();
  if (tid < NUM_GRAPHS && h[tid] != 0.f) atomicAdd(&counts[tid], h[tid]);
}

__device__ __forceinline__ void cd_fill4(v16bf& a, int base, float4 f, float& psq) {
  a[base + 0] = (__bf16)f.x; a[base + 1] = (__bf16)f.y;
  a[base + 2] = (__bf16)f.z; a[base + 3] = (__bf16)f.w;
  psq += f.x * f.x + f.y * f.y + f.z * f.z + f.w * f.w;
}

// load one column's 4 K-step B fragments from LDS (ISA B layout: half-wave holds
// K in [8h,8h+8) and [16+8h,16+8h+8) of its column)
__device__ __forceinline__ void cd_loadB(const __bf16* __restrict__ crow, int h,
                                         v16bf (&B)[4]) {
  #pragma unroll
  for (int k = 0; k < 4; ++k) {
    v8bf lo = *(const v8bf*)(crow + k * 32 + 8 * h);
    v8bf hi = *(const v8bf*)(crow + k * 32 + 16 + 8 * h);
    #pragma unroll
    for (int i = 0; i < 8; ++i) { B[k][i] = lo[i]; B[k][i + 8] = hi[i]; }
  }
}

// -------- main: WMMA distance tile + sorted-segment accumulation --------
// block = 256 threads (8 waves), 128 rows/block, all 256 centroids.
__global__ __launch_bounds__(256) void cd_main_kernel(const float* __restrict__ x,
                                                      const int* __restrict__ batch,
                                                      const unsigned short* __restrict__ centBf,
                                                      const float* __restrict__ cSq,
                                                      float* __restrict__ sums) {
  extern __shared__ char smem[];
  __bf16* sCent  = (__bf16*)smem;                           // 65536 B
  float*  sAcc   = (float*)(smem + 65536);                  //  8192 B
  float*  sCsq   = (float*)(smem + 65536 + 8192);           //  1024 B
  int*    sBatch = (int*)(smem + 65536 + 8192 + 1024);      //   512 B

  const int tid = threadIdx.x;

  // ---- block preload ----
  {
    const uint4* src = (const uint4*)centBf;   // 64KB table, L2 resident
    uint4* dst = (uint4*)sCent;
    #pragma unroll 4
    for (int i = tid; i < 4096; i += 256) dst[i] = src[i];
    #pragma unroll
    for (int i = tid; i < NSEG * 256; i += 256) sAcc[i] = 0.f;
    sCsq[tid] = cSq[tid];
    if (tid < 128) sBatch[tid] = batch[blockIdx.x * 128 + tid];
  }
  __syncthreads();

  const int wave = tid >> 5, lane = tid & 31;
  const int m = lane & 15, h = lane >> 4;       // A row / D col = m ; half = h
  const int rowBase = blockIdx.x * 128 + wave * 16;

  // block-uniform segment info -> scalar registers, scalar branches
  const int gBase   = __builtin_amdgcn_readfirstlane(sBatch[0]);
  const int nsegBlk = __builtin_amdgcn_readfirstlane(sBatch[127]) - gBase + 1;
  const bool fastSeg = (nsegBlk <= NSEG);

  // ---- A fragments (16 rows x K=128 bf16) + row norms in f32 ----
  // ISA layout: lanes 0-15 hold K in [8h,8h+8) and [16+8h,16+8h+8) per 32-K step.
  v16bf A[4];
  float psq = 0.f;
  const float* xrow = x + (size_t)(rowBase + m) * EMBED_DIM;
  #pragma unroll
  for (int k = 0; k < 4; ++k) {
    const float* p0 = xrow + k * 32 + 8 * h;
    const float* p1 = xrow + k * 32 + 16 + 8 * h;
    cd_fill4(A[k], 0,  *(const float4*)(p0),     psq);
    cd_fill4(A[k], 4,  *(const float4*)(p0 + 4), psq);
    cd_fill4(A[k], 8,  *(const float4*)(p1),     psq);
    cd_fill4(A[k], 12, *(const float4*)(p1 + 4), psq);
  }
  // lanes m and m+16 each hold half of row m's elements
  float xsqFull = psq + __shfl_xor(psq, 16, 32);

  // D-matrix: lane holds rows m = v + 8h (v=0..7), col = lane%16
  float xs[8]; int sg[8];
  #pragma unroll
  for (int v = 0; v < 8; ++v) {
    xs[v] = __shfl(xsqFull, v + 8 * h, 32);
    sg[v] = sBatch[wave * 16 + v + 8 * h] - gBase;   // non-decreasing in v
  }

  // ---- 16 column tiles of 16 centroids, K=128, double-buffered B ----
  v16bf B0[4], B1[4];
  cd_loadB(sCent + (size_t)m * EMBED_DIM, h, B0);

  #pragma unroll 2
  for (int ct = 0; ct < 16; ++ct) {
    const int cur = ct & 1;                      // constant after 2x unroll
    const int col = ct * 16 + m;
    v16bf (&Bc)[4] = cur ? B1 : B0;
    v16bf (&Bn)[4] = cur ? B0 : B1;
    if (ct < 15) cd_loadB(sCent + (size_t)(col + 16) * EMBED_DIM, h, Bn);

    v8f acc = {0.f, 0.f, 0.f, 0.f, 0.f, 0.f, 0.f, 0.f};
    #pragma unroll
    for (int k = 0; k < 4; ++k)
      acc = __builtin_amdgcn_wmma_f32_16x16x32_bf16(
          false, A[k], false, Bc[k], (short)0, acc, false, false);

    const float cs = sCsq[col];
    float dist[8];
    #pragma unroll
    for (int v = 0; v < 8; ++v) {
      float d2 = xs[v] + cs - 2.0f * acc[v];
      dist[v] = __builtin_amdgcn_sqrtf(fmaxf(d2, 0.f));   // raw v_sqrt_f32
    }

    if (fastSeg) {                               // scalar branch (readfirstlane)
      if (sg[0] == sg[7]) {                      // all 8 rows in one graph (hot path)
        float run = ((dist[0] + dist[1]) + (dist[2] + dist[3])) +
                    ((dist[4] + dist[5]) + (dist[6] + dist[7]));
        atomicAdd(&sAcc[sg[0] * 256 + col], run);          // 1 ds_add_f32
      } else {                                   // graph boundary inside lane's rows
        #pragma unroll
        for (int v = 0; v < 8; ++v) atomicAdd(&sAcc[sg[v] * 256 + col], dist[v]);
      }
    } else {                                     // pathological span: direct global
      #pragma unroll
      for (int v = 0; v < 8; ++v)
        atomicAdd(&sums[(size_t)(gBase + sg[v]) * 256 + col], dist[v]);
    }
  }

  // ---- flush LDS segment sums (batch sorted -> nseg is tiny) ----
  __syncthreads();
  if (fastSeg) {
    for (int i = tid; i < nsegBlk * 256; i += 256)
      atomicAdd(&sums[(size_t)(gBase + (i >> 8)) * 256 + (i & 255)], sAcc[i]);
  }
}

// -------- finalize: mean --------
__global__ __launch_bounds__(256) void cd_final_kernel(const float* __restrict__ sums,
                                                       const float* __restrict__ counts,
                                                       float* __restrict__ out) {
  int g = blockIdx.x, c = threadIdx.x;
  out[g * NUM_CENTROID + c] = sums[g * NUM_CENTROID + c] / fmaxf(counts[g], 1.f);
}

extern "C" void kernel_launch(void* const* d_in, const int* in_sizes, int n_in,
                              void* d_out, int out_size, void* d_ws, size_t ws_size,
                              hipStream_t stream) {
  const float* x     = (const float*)d_in[0];
  const int*   batch = (const int*)d_in[1];
  const float* cw    = (const float*)d_in[2];

  char* ws = (char*)d_ws;
  unsigned short* centBf = (unsigned short*)ws;
  float* cSq    = (float*)(ws + WS_CSQ);
  float* sums   = (float*)(ws + WS_SUMS);
  float* counts = (float*)(ws + WS_COUNTS);
  float* out    = (float*)d_out;

  cd_zero_kernel<<<(NUM_GRAPHS * NUM_CENTROID + 255) / 256, 256, 0, stream>>>(sums, counts);
  cd_cent_kernel<<<NUM_CENTROID, 128, 0, stream>>>(cw, centBf, cSq);
  cd_hist_kernel<<<NUM_GRAPHS, 256, 0, stream>>>(batch, counts);

  size_t smemBytes = 65536 + 8192 + 1024 + 512;   // cent + segAccum + csq + batch
  cd_main_kernel<<<N_NODES / 128, 256, smemBytes, stream>>>(x, batch, centBf, cSq, sums);

  cd_final_kernel<<<NUM_GRAPHS, 256, 0, stream>>>(sums, counts, out);
}
